// RPN_63058709840282
// MI455X (gfx1250) — compile-verified, matrix operand order checked by reading, not run on hardware
//
#include <hip/hip_runtime.h>
#include <hip/hip_bf16.h>
#include <math.h>
#include <stdint.h>

typedef __attribute__((ext_vector_type(16))) _Float16 v16h;
typedef __attribute__((ext_vector_type(8)))  float    v8f;
typedef __attribute__((ext_vector_type(4)))  unsigned int v4u;
typedef __attribute__((ext_vector_type(8)))  int      v8i;
typedef __attribute__((ext_vector_type(4)))  int      v4i;

#define CIN   512
#define COUT  512
#define HF    96
#define NPIX  (HF*HF)          /* 9216 */
#define KDIM  (CIN*9)          /* 4608 */
#define NA    (NPIX*9)         /* 82944 */
#define NTOP  6000
#define NW    ((NTOP+31)/32)   /* 188 */
#define NBINS 8192
#define IMGSZ 1536.0f

#define TM 128
#define TN 128
#define KT 32
#define LSTR 40                 /* halves per LDS row: 64B payload + 16B pad (TDM pad 4 DW / 16 DW) */

#define HAVE_TDM __has_builtin(__builtin_amdgcn_tensor_load_to_lds)

union V16U { v16h h; uint4 u[2]; };

/* ---------------- weight pack: conv_w (O,I,3,3) f32 -> Aw[m][k] f16, k = r*512+i ---------------- */
__global__ void pack_weights(const float* __restrict__ cw, _Float16* __restrict__ Aw) {
  int idx = blockIdx.x * 256 + threadIdx.x;
  if (idx >= COUT * KDIM) return;
  int m = idx / KDIM;
  int k = idx - m * KDIM;
  int r = k / CIN;
  int i = k - r * CIN;
  Aw[idx] = (_Float16)cw[((size_t)m * CIN + i) * 9 + r];
}

/* ---------------- 3x3 conv as implicit-GEMM WMMA f16, A tile via TDM ---------------- */
__global__ __launch_bounds__(256) void conv3x3_wmma(
    const _Float16* __restrict__ Aw, const float* __restrict__ feat,
    const float* __restrict__ bias, float* __restrict__ xout) {
  __shared__ _Float16 Als[TM * LSTR];
  __shared__ _Float16 Bls[TN * LSTR];
  const int tid  = threadIdx.x;
  const int lane = tid & 31;
  const int wave = tid >> 5;
  const int wm   = wave & 3;   /* 4 waves along M (32 rows each)  */
  const int wn   = wave >> 2;  /* 2 waves along N (64 cols each)  */
  const int mblk = blockIdx.y * TM;
  const int nblk = blockIdx.x * TN;

  v8f zero = {};
  v8f acc[2][4];
#pragma unroll
  for (int i = 0; i < 2; i++)
#pragma unroll
    for (int j = 0; j < 4; j++) acc[i][j] = zero;

  const int lrow = tid & 127;       /* local m or n row for loaders */
  const int k16  = (tid >> 7) << 4; /* 0 or 16: half of the K window */
  const int nb   = nblk + lrow;
  const int by   = nb / HF;
  const int bxp  = nb - by * HF;

#if HAVE_TDM
  const unsigned ldsA = (unsigned)(uintptr_t)(void*)&Als[0]; /* byte offset in workgroup LDS */
  const bool tdm_wave = (tid < 32);
#endif

  for (int r = 0; r < 9; r++) {
    /* halo geometry: loop-invariant per filter tap */
    const int dy = r / 3 - 1, dx = r - (r / 3) * 3 - 1;
    const int yy = by + dy, xx = bxp + dx;
    const bool inb = (yy >= 0) && (yy < HF) && (xx >= 0) && (xx < HF);
    const int yyc = yy < 0 ? 0 : (yy > HF - 1 ? HF - 1 : yy);
    const int xxc = xx < 0 ? 0 : (xx > HF - 1 ? HF - 1 : xx);
    const float* fbase = feat + (size_t)k16 * NPIX + yyc * HF + xxc;

    for (int ks = 0; ks < 16; ks++) {
      const int kt = r * 512 + ks * 32;

      /* ---- A tile: Tensor Data Mover 2-D tile, 128 rows x 64B, LDS pad 16B/row ---- */
#if HAVE_TDM
      if (tdm_wave) {
        unsigned long long ga =
            (unsigned long long)(uintptr_t)(Aw + (size_t)mblk * KDIM + kt);
        v4u g0;
        g0[0] = 1u;                                   /* count=1, user mode */
        g0[1] = ldsA;                                 /* lds_addr */
        g0[2] = (unsigned)(ga & 0xFFFFFFFFu);         /* global_addr[31:0] */
        g0[3] = ((unsigned)(ga >> 32) & 0x01FFFFFFu)  /* global_addr[56:32] */
                | 0x80000000u;                        /* type=2 (image) */
        v8i g1;
        g1[0] = (1 << 16)      /* data_size = 2B */
              | (1 << 20)      /* pad_enable */
              | (3 << 22)      /* pad_interval: 16 DWORDs */
              | (3 << 25);     /* pad_amount: 4 DWORDs */
        g1[1] = (int)((KDIM & 0xFFFF) << 16);                 /* tensor_dim0[15:0] */
        g1[2] = (int)((KDIM >> 16) | ((CIN & 0xFFFF) << 16)); /* td0 hi | tensor_dim1 lo */
        g1[3] = (int)((CIN >> 16) | (32 << 16));              /* td1 hi | tile_dim0=32 */
        g1[4] = 128;                                          /* tile_dim1=128, tile_dim2=0 */
        g1[5] = KDIM;                                         /* tensor_dim0_stride[31:0] */
        g1[6] = 0;
        g1[7] = 0;
        v4i gz4 = {0, 0, 0, 0};
        v8i gz8 = {0, 0, 0, 0, 0, 0, 0, 0};
        __builtin_amdgcn_tensor_load_to_lds(g0, g1, gz4, gz4, gz8, 0);
      }
#else
      {
        const _Float16* src = Aw + (size_t)(mblk + lrow) * KDIM + (kt + k16);
        uint4 d0 = *(const uint4*)src;
        uint4 d1 = *(const uint4*)(src + 8);
        *(uint4*)&Als[lrow * LSTR + k16]     = d0;
        *(uint4*)&Als[lrow * LSTR + k16 + 8] = d1;
      }
#endif

      /* ---- B tile: branchless im2col (clamped addresses, value select) ---- */
      {
        const float* fs = fbase + (size_t)(ks * 32) * NPIX;
        _Float16 tmp[16];
#pragma unroll
        for (int e = 0; e < 16; e++) {
          float v = fs[(size_t)e * NPIX];
          tmp[e] = inb ? (_Float16)v : (_Float16)0.0f;
        }
        *(uint4*)&Bls[lrow * LSTR + k16]     = *(const uint4*)&tmp[0];
        *(uint4*)&Bls[lrow * LSTR + k16 + 8] = *(const uint4*)&tmp[8];
      }

#if HAVE_TDM
      if (tdm_wave) __builtin_amdgcn_s_wait_tensorcnt((short)0);
#endif
      __syncthreads();

      /* fragments per documented 16-bit A 16x32 layout */
      const int h0 = (lane >> 4) << 3;
      const int fr = lane & 15;
      v16h af[2], bf[4];
#pragma unroll
      for (int i = 0; i < 2; i++) {
        V16U u;
        const int row = wm * 32 + i * 16 + fr;
        u.u[0] = *(const uint4*)&Als[row * LSTR + h0];
        u.u[1] = *(const uint4*)&Als[row * LSTR + h0 + 16];
        af[i] = u.h;
      }
#pragma unroll
      for (int j = 0; j < 4; j++) {
        V16U u;
        const int col = wn * 64 + j * 16 + fr;
        u.u[0] = *(const uint4*)&Bls[col * LSTR + h0];
        u.u[1] = *(const uint4*)&Bls[col * LSTR + h0 + 16];
        bf[j] = u.h;
      }
#pragma unroll
      for (int i = 0; i < 2; i++)
#pragma unroll
        for (int j = 0; j < 4; j++)
          acc[i][j] = __builtin_amdgcn_wmma_f32_16x16x32_f16(
              false, af[i], false, bf[j], (short)0, acc[i][j], false, false);
      __syncthreads();
    }
  }

  /* epilogue: bias + relu; C layout: VGPR v -> M = v (lanes 0-15) / v+8 (lanes 16-31) */
  const int fr = lane & 15;
  const int hi = lane >> 4;
#pragma unroll
  for (int i = 0; i < 2; i++) {
#pragma unroll
    for (int j = 0; j < 4; j++) {
      const int col = nblk + wn * 64 + j * 16 + fr;
#pragma unroll
      for (int v = 0; v < 8; v++) {
        const int row = mblk + wm * 32 + i * 16 + hi * 8 + v;
        float val = acc[i][j][v] + bias[row];
        xout[(size_t)row * NPIX + col] = fmaxf(val, 0.0f);
      }
    }
  }
}

/* ---------------- 1x1 heads + softmax + score max ---------------- */
__global__ __launch_bounds__(256) void heads(
    const float* __restrict__ x,
    const float* __restrict__ tw, const float* __restrict__ tb,
    const float* __restrict__ bw, const float* __restrict__ bb,
    float* __restrict__ prop, float* __restrict__ scores, float* __restrict__ sArr) {
  int n = blockIdx.x * 256 + threadIdx.x;
  if (n >= NPIX) return;
  float st[18], bxv[36];
#pragma unroll
  for (int o = 0; o < 18; o++) st[o] = tb[o];
#pragma unroll
  for (int o = 0; o < 36; o++) bxv[o] = bb[o];
  for (int c = 0; c < CIN; c++) {
    float xv = x[(size_t)c * NPIX + n];
#pragma unroll
    for (int o = 0; o < 18; o++) st[o] = fmaf(xv, tw[o * CIN + c], st[o]);
#pragma unroll
    for (int o = 0; o < 36; o++) bxv[o] = fmaf(xv, bw[o * CIN + c], bxv[o]);
  }
#pragma unroll
  for (int a = 0; a < 9; a++) {
    float p0 = st[2 * a], p1 = st[2 * a + 1];
    float m = fmaxf(p0, p1);
    float e0 = expf(p0 - m), e1 = expf(p1 - m);
    float inv = 1.0f / (e0 + e1);
    float q0 = e0 * inv, q1 = e1 * inv;
    int row = n * 9 + a;
    scores[2 * row] = q0;
    scores[2 * row + 1] = q1;
    sArr[row] = fmaxf(q0, q1);
#pragma unroll
    for (int c = 0; c < 4; c++) prop[row * 4 + c] = bxv[4 * a + c];
  }
}

/* ---------------- anchors (faithful formulas incl. swapped voc2yolo), IoU vs gt, targets ---------------- */
__global__ void anchors_iou_tgt(const float* __restrict__ bbx,
                                float* __restrict__ anchors,
                                float* __restrict__ iouA,
                                int* __restrict__ tgtA) {
  int idx = blockIdx.x * 256 + threadIdx.x;
  if (idx >= NA) return;
  int pix = idx / 9, j = idx - pix * 9;
  int py = pix / HF, px = pix - py * HF;
  const float stride = IMGSZ / (float)HF;
  int ridx = j / 3, sidx = j - ridx * 3;
  const float ratios[3] = {0.5f, 1.0f, 2.0f};
  const float scales[3] = {8.0f, 16.0f, 32.0f};
  float b2 = stride - 1.0f, b3 = stride - 1.0f;
  float ya0 = 0.5f * (b3 - 1.0f), ya1 = 0.5f * (b2 - 1.0f);
  float ya2 = b3 + 1.0f, ya3 = b2 + 1.0f;
  float wr = rintf(sqrtf(ya2 * ya3 / ratios[ridx]));
  float hr = rintf(wr * ratios[ridx]);
  float a0 = ya0 - 0.5f * (wr - 1.0f), a1 = ya1 - 0.5f * (hr - 1.0f);
  float a2 = ya0 + 0.5f * (wr - 1.0f), a3 = ya1 + 0.5f * (hr - 1.0f);
  float y0 = a0 + 0.5f * (a3 - 1.0f), y1 = a1 + 0.5f * (a2 - 1.0f);
  float y2 = a3 - a1 + 1.0f, y3 = a2 - a0 + 1.0f;
  float sc = scales[sidx];
  float w = y2 * sc, h = y3 * sc;
  float x1 = y0 - 0.5f * (w - 1.0f) + px * stride;
  float yy1 = y1 - 0.5f * (h - 1.0f) + py * stride;
  float x2 = y0 + 0.5f * (w - 1.0f) + px * stride;
  float yy2 = y1 + 0.5f * (h - 1.0f) + py * stride;
  x1 = fminf(fmaxf(x1, 0.f), IMGSZ);
  yy1 = fminf(fmaxf(yy1, 0.f), IMGSZ);
  x2 = fminf(fmaxf(x2, 0.f), IMGSZ);
  yy2 = fminf(fmaxf(yy2, 0.f), IMGSZ);
  anchors[idx * 4 + 0] = x1;
  anchors[idx * 4 + 1] = yy1;
  anchors[idx * 4 + 2] = x2;
  anchors[idx * 4 + 3] = yy2;
  float g0 = bbx[0], g1 = bbx[1], g2 = bbx[2], g3 = bbx[3];
  float areaG = (g2 - g0) * (g3 - g1);
  float areaA = (x2 - x1) * (yy2 - yy1);
  float lx = fmaxf(g0, x1), ly = fmaxf(g1, yy1);
  float rx = fminf(g2, x2), ry = fminf(g3, yy2);
  float iw = fmaxf(rx - lx, 0.f), ih = fmaxf(ry - ly, 0.f);
  float inter = iw * ih;
  float iou = inter / (areaG + areaA - inter);
  iouA[idx] = iou;
  int t = -1;
  if (iou >= 0.5f) t = 1;
  if (iou <= 0.1f) t = 0;
  tgtA[idx] = t;
}

/* ---------------- selection: histogram -> threshold bin -> exact stable ranks (ascending) ---------------- */
__global__ void zero_hist(int* __restrict__ hist) {
  int i = blockIdx.x * 256 + threadIdx.x;
  if (i < NBINS) hist[i] = 0;
}

__global__ void hist_k(const float* __restrict__ s, int* __restrict__ hist) {
  int i = blockIdx.x * 256 + threadIdx.x;
  float v = s[i];
  int b = (int)(v * (float)NBINS);
  b = b < 0 ? 0 : (b > NBINS - 1 ? NBINS - 1 : b);
  atomicAdd(&hist[b], 1);
}

__global__ void find_thresh(const int* __restrict__ hist, int* __restrict__ thr) {
  if (threadIdx.x == 0 && blockIdx.x == 0) {
    int cum = 0, B = NBINS - 1;
    for (int b = 0; b < NBINS; b++) {
      cum += hist[b];
      if (cum >= NTOP) { B = b; break; }
    }
    *thr = B;
  }
}

__global__ __launch_bounds__(256) void rank_select(const float* __restrict__ s,
                                                   const int* __restrict__ thr,
                                                   int* __restrict__ top_idx) {
  __shared__ float tile[256];
  const int i = blockIdx.x * 256 + threadIdx.x;
  const float si = s[i];
  int b = (int)(si * (float)NBINS);
  b = b < 0 ? 0 : (b > NBINS - 1 ? NBINS - 1 : b);
  const int B = *thr;
  const bool cand = (b <= B);
  int rank = 0;
  for (int base = 0; base < NA; base += 256) {
    tile[threadIdx.x] = s[base + threadIdx.x];
    __syncthreads();
    if (cand) {
#pragma unroll 8
      for (int t = 0; t < 256; t++) {
        float sj = tile[t];
        int jj = base + t;
        rank += ((sj < si) || (sj == si && jj < i)) ? 1 : 0;
      }
    }
    __syncthreads();
  }
  if (cand && rank < NTOP) top_idx[rank] = i;
}

__global__ void gather_top(const int* __restrict__ top_idx, const float* __restrict__ iouA,
                           const float* __restrict__ anchors,
                           float* __restrict__ topS, float* __restrict__ topBox) {
  int k = blockIdx.x * 256 + threadIdx.x;
  if (k >= NTOP) return;
  int idx = top_idx[k];
  topS[k] = iouA[idx];
  topBox[k * 4 + 0] = anchors[idx * 4 + 0];
  topBox[k * 4 + 1] = anchors[idx * 4 + 1];
  topBox[k * 4 + 2] = anchors[idx * 4 + 2];
  topBox[k * 4 + 3] = anchors[idx * 4 + 3];
}

/* stable descending argsort by IoU score via exact ranks; also gathers sorted boxes */
__global__ __launch_bounds__(256) void order_desc(const float* __restrict__ topS,
                                                  const float* __restrict__ topBox,
                                                  int* __restrict__ order,
                                                  float* __restrict__ sb) {
  __shared__ float tile[256];
  int k = blockIdx.x * 256 + threadIdx.x;
  float vk = (k < NTOP) ? topS[k] : -3.0e38f;
  int rank = 0;
  for (int base = 0; base < NTOP; base += 256) {
    int j = base + threadIdx.x;
    tile[threadIdx.x] = (j < NTOP) ? topS[j] : -3.0e38f;
    __syncthreads();
    int lim = (NTOP - base) < 256 ? (NTOP - base) : 256;
    if (k < NTOP) {
      for (int t = 0; t < lim; t++) {
        float vj = tile[t];
        int jj = base + t;
        rank += ((vj > vk) || (vj == vk && jj < k)) ? 1 : 0;
      }
    }
    __syncthreads();
  }
  if (k < NTOP) {
    order[rank] = k;
    sb[rank * 4 + 0] = topBox[k * 4 + 0];
    sb[rank * 4 + 1] = topBox[k * 4 + 1];
    sb[rank * 4 + 2] = topBox[k * 4 + 2];
    sb[rank * 4 + 3] = topBox[k * 4 + 3];
  }
}

/* ---------------- greedy NMS: suppression bitmask + serial scan ---------------- */
__global__ void nms_mask(const float* __restrict__ sb, unsigned* __restrict__ mask) {
  int t = blockIdx.x * 256 + threadIdx.x;
  if (t >= NTOP * NW) return;
  int row = t / NW, w = t - row * NW;
  float r0 = sb[row * 4], r1 = sb[row * 4 + 1], r2 = sb[row * 4 + 2], r3 = sb[row * 4 + 3];
  float areaR = (r2 - r0) * (r3 - r1);
  unsigned bits = 0u;
  for (int b = 0; b < 32; b++) {
    int col = w * 32 + b;
    if (col < NTOP && col > row) {
      float c0 = sb[col * 4], c1 = sb[col * 4 + 1], c2 = sb[col * 4 + 2], c3 = sb[col * 4 + 3];
      float areaC = (c2 - c0) * (c3 - c1);
      float lx = fmaxf(r0, c0), ly = fmaxf(r1, c1);
      float rx = fminf(r2, c2), ry = fminf(r3, c3);
      float iw = fmaxf(rx - lx, 0.f), ih = fmaxf(ry - ly, 0.f);
      float inter = iw * ih;
      float iou = inter / (areaR + areaC - inter);
      if (iou > 0.6f) bits |= (1u << b);
    }
  }
  mask[t] = bits;
}

__global__ __launch_bounds__(256) void nms_seq(const unsigned* __restrict__ mask,
                                               int* __restrict__ keep) {
  __shared__ unsigned removed[NW];
  __shared__ int flag;
  int tid = threadIdx.x;
  if (tid < NW) removed[tid] = 0u;
  __syncthreads();
  for (int i = 0; i < NTOP; i++) {
    if (tid == 0) {
      int f = !((removed[i >> 5] >> (i & 31)) & 1u);
      flag = f;
      keep[i] = f;
    }
    __syncthreads();
    if (flag && tid < NW) removed[tid] |= mask[(size_t)i * NW + tid];
    __syncthreads();
  }
}

/* ---------------- stable partition (kept first), bug-faithful roi gather, outputs ---------------- */
__global__ __launch_bounds__(1024) void final_sel(const int* __restrict__ keep,
                                                  const int* __restrict__ order,
                                                  const float* __restrict__ anchors,
                                                  const int* __restrict__ tgtA,
                                                  float* __restrict__ out) {
  __shared__ unsigned char kp[NTOP];
  __shared__ int totalk;
  int tid = threadIdx.x;
  for (int i = tid; i < NTOP; i += 1024) kp[i] = (unsigned char)keep[i];
  __syncthreads();
  if (tid == 0) {
    int s = 0;
    for (int i = 0; i < NTOP; i++) s += kp[i];
    totalk = s;
  }
  __syncthreads();
  float* roi   = out + (size_t)NA * 4;
  float* roit  = roi + NTOP * 4;
  float* keepf = roit + NTOP;
  int tot = totalk;
  for (int i = tid; i < NTOP; i += 1024) {
    int pre = 0;
    for (int j = 0; j < i; j++) pre += kp[j];
    int k = kp[i];
    int slot = k ? pre : tot + (i - pre);
    int oi = order[i]; /* bug-faithful: index in [0,6000) into FULL anchor set */
    if (k) {
      roi[slot * 4 + 0] = anchors[oi * 4 + 0];
      roi[slot * 4 + 1] = anchors[oi * 4 + 1];
      roi[slot * 4 + 2] = anchors[oi * 4 + 2];
      roi[slot * 4 + 3] = anchors[oi * 4 + 3];
      roit[slot] = (float)tgtA[oi];
    } else {
      roi[slot * 4 + 0] = 0.f;
      roi[slot * 4 + 1] = 0.f;
      roi[slot * 4 + 2] = 0.f;
      roi[slot * 4 + 3] = 0.f;
      roit[slot] = -1.0f;
    }
    keepf[slot] = (float)k;
  }
}

/* ---------------- losses: deterministic two-stage reduction ---------------- */
__global__ __launch_bounds__(256) void losses_partial(const float* __restrict__ prop,
                                                      const float* __restrict__ anchors,
                                                      const int* __restrict__ tgtA,
                                                      const float* __restrict__ scores,
                                                      float* __restrict__ partials) {
  int i = blockIdx.x * 256 + threadIdx.x;
  float bsum = 0.f, pcnt = 0.f, csum = 0.f, vcnt = 0.f;
  int t = tgtA[i];
  if (t > 0) {
#pragma unroll
    for (int c = 0; c < 4; c++) {
      float d = prop[i * 4 + c] - anchors[i * 4 + c];
      float ad = fabsf(d);
      bsum += (ad < 1.0f) ? 0.5f * d * d : ad - 0.5f;
    }
    pcnt = 1.f;
  }
  if (t >= 0) {
    int tc = t > 1 ? 1 : t;
    float q0 = scores[2 * i], q1 = scores[2 * i + 1];
    float m = fmaxf(q0, q1);
    float lse = m + logf(expf(q0 - m) + expf(q1 - m));
    float qt = tc ? q1 : q0;
    csum = -(qt - lse);
    vcnt = 1.f;
  }
  __shared__ float red[256];
  float vals[4] = {bsum, pcnt, csum, vcnt};
#pragma unroll
  for (int v = 0; v < 4; v++) {
    red[threadIdx.x] = vals[v];
    __syncthreads();
    for (int s = 128; s > 0; s >>= 1) {
      if (threadIdx.x < s) red[threadIdx.x] += red[threadIdx.x + s];
      __syncthreads();
    }
    if (threadIdx.x == 0) partials[blockIdx.x * 4 + v] = red[0];
    __syncthreads();
  }
}

__global__ void finalize(const float* __restrict__ partials, float* __restrict__ out) {
  if (threadIdx.x != 0 || blockIdx.x != 0) return;
  float bsum = 0.f, pcnt = 0.f, csum = 0.f, vcnt = 0.f;
  for (int b = 0; b < NA / 256; b++) {
    bsum += partials[b * 4 + 0];
    pcnt += partials[b * 4 + 1];
    csum += partials[b * 4 + 2];
    vcnt += partials[b * 4 + 3];
  }
  size_t base = (size_t)NA * 4 + (size_t)NTOP * 4 + NTOP + NTOP;
  out[base + 0] = bsum / fmaxf(pcnt * 4.0f, 1.0f);
  out[base + 1] = csum / fmaxf(vcnt, 1.0f);
}

/* ---------------- launch ---------------- */
extern "C" void kernel_launch(void* const* d_in, const int* in_sizes, int n_in,
                              void* d_out, int out_size, void* d_ws, size_t ws_size,
                              hipStream_t stream) {
  (void)in_sizes; (void)n_in; (void)out_size; (void)ws_size;
  const float* features = (const float*)d_in[1];
  const float* bbx      = (const float*)d_in[2];
  const float* conv_w   = (const float*)d_in[4];
  const float* conv_b   = (const float*)d_in[5];
  const float* tgt_w    = (const float*)d_in[6];
  const float* tgt_b    = (const float*)d_in[7];
  const float* bx_w     = (const float*)d_in[8];
  const float* bx_b     = (const float*)d_in[9];
  float* out = (float*)d_out;
  char* ws = (char*)d_ws;

  size_t off = 0;
  _Float16* Aw = (_Float16*)(ws + off); off += (size_t)COUT * KDIM * 2;
  float* xbuf  = (float*)(ws + off);    off += (size_t)COUT * NPIX * 4;
  float* scr   = (float*)(ws + off);    off += (size_t)NA * 2 * 4;
  float* sArr  = (float*)(ws + off);    off += (size_t)NA * 4;
  float* anch  = (float*)(ws + off);    off += (size_t)NA * 4 * 4;
  float* iouA  = (float*)(ws + off);    off += (size_t)NA * 4;
  int*   tgtA  = (int*)(ws + off);      off += (size_t)NA * 4;
  int*   hist  = (int*)(ws + off);      off += (size_t)NBINS * 4;
  int*   thr   = (int*)(ws + off);      off += 256;
  int*   topI  = (int*)(ws + off);      off += (size_t)NTOP * 4;
  float* topS  = (float*)(ws + off);    off += (size_t)NTOP * 4;
  float* topB  = (float*)(ws + off);    off += (size_t)NTOP * 4 * 4;
  int*   ordA  = (int*)(ws + off);      off += (size_t)NTOP * 4;
  float* sb    = (float*)(ws + off);    off += (size_t)NTOP * 4 * 4;
  unsigned* msk = (unsigned*)(ws + off); off += (size_t)NTOP * NW * 4;
  int*   keepA = (int*)(ws + off);      off += (size_t)NTOP * 4;
  float* part  = (float*)(ws + off);    off += (size_t)(NA / 256) * 4 * 4;

  pack_weights<<<(COUT * KDIM + 255) / 256, 256, 0, stream>>>(conv_w, Aw);
  conv3x3_wmma<<<dim3(NPIX / TN, COUT / TM), 256, 0, stream>>>(Aw, features, conv_b, xbuf);
  heads<<<(NPIX + 255) / 256, 256, 0, stream>>>(xbuf, tgt_w, tgt_b, bx_w, bx_b, out, scr, sArr);
  anchors_iou_tgt<<<(NA + 255) / 256, 256, 0, stream>>>(bbx, anch, iouA, tgtA);
  zero_hist<<<NBINS / 256, 256, 0, stream>>>(hist);
  hist_k<<<NA / 256, 256, 0, stream>>>(sArr, hist);
  find_thresh<<<1, 1, 0, stream>>>(hist, thr);
  rank_select<<<NA / 256, 256, 0, stream>>>(sArr, thr, topI);
  gather_top<<<(NTOP + 255) / 256, 256, 0, stream>>>(topI, iouA, anch, topS, topB);
  order_desc<<<(NTOP + 255) / 256, 256, 0, stream>>>(topS, topB, ordA, sb);
  nms_mask<<<(NTOP * NW + 255) / 256, 256, 0, stream>>>(sb, msk);
  nms_seq<<<1, 256, 0, stream>>>(msk, keepA);
  final_sel<<<1, 1024, 0, stream>>>(keepA, ordA, anch, tgtA, out);
  losses_partial<<<NA / 256, 256, 0, stream>>>(out, anch, tgtA, scr, part);
  finalize<<<1, 1, 0, stream>>>(part, out);
}
